// DecoupledMemoryAttnLayer_34402688041107
// MI455X (gfx1250) — compile-verified, matrix operand order checked by reading, not run on hardware
//
#include <hip/hip_runtime.h>
#include <math.h>

#define NQ 4096
#define NKK 16448
#define CC 256
#define FF 2048
#define LN_EPS 1e-5f
#define SA_SPLIT 4
#define CA_SPLIT 8

typedef __attribute__((ext_vector_type(16))) __bf16 v16bf;
typedef __attribute__((ext_vector_type(8)))  __bf16 v8bf;
typedef __attribute__((ext_vector_type(4)))  __bf16 v4bf;
typedef __attribute__((ext_vector_type(8)))  float  v8f;

// ---------------------------------------------------------------------------
// WMMA helpers (CDNA5: D = A(16x32 bf16) x B(32x16 bf16) + C(16x16 f32))
// ---------------------------------------------------------------------------
__device__ __forceinline__ v8f wmma_bf16(v16bf a, v16bf b, v8f c) {
  return __builtin_amdgcn_wmma_f32_16x16x32_bf16(
      /*neg_a=*/false, a, /*neg_b=*/false, b,
      /*c_mod=*/(short)0, c, /*reuse_a=*/false, /*reuse_b=*/false);
}

// 16-bit A/B fragment per ISA layout (16x32 MxK): lane owns row (lane&15),
// lanes>=16 start at K offset 8; elements 0..7 <- K=koff+0..7,
// elements 8..15 <- K=koff+16..23 (two contiguous 16B loads).
// Works for both global and LDS pointers (rows contiguous).
__device__ __forceinline__ v16bf load_frag(const __bf16* p) {
  v8bf lo = *(const v8bf*)(p);
  v8bf hi = *(const v8bf*)(p + 16);
  v16bf r;
#pragma unroll
  for (int i = 0; i < 8; ++i) { r[i] = lo[i]; r[8 + i] = hi[i]; }
  return r;
}

// ---------------------------------------------------------------------------
// Elementwise converts (4-wide: float4 load, 8B bf16 store)
// ---------------------------------------------------------------------------
__global__ void cvt_bf16_kernel(const float* __restrict__ in,
                                __bf16* __restrict__ out, int n) {
  int i = (blockIdx.x * 256 + threadIdx.x) * 4;
  if (i < n) {
    float4 v = *(const float4*)(in + i);
    v4bf o;
    o[0] = (__bf16)v.x; o[1] = (__bf16)v.y;
    o[2] = (__bf16)v.z; o[3] = (__bf16)v.w;
    *(v4bf*)(out + i) = o;
  }
}

// fp32 [M][N] -> bf16 [N][M], 32x32 LDS tiles, both sides coalesced
__global__ __launch_bounds__(256)
void transpose_cvt_kernel(const float* __restrict__ in,
                          __bf16* __restrict__ out, int M, int N) {
  __shared__ float tile[32][33];
  int tx = threadIdx.x & 31, ty = threadIdx.x >> 5;
  int bn = blockIdx.x * 32, bm = blockIdx.y * 32;
#pragma unroll
  for (int j = 0; j < 32; j += 8)
    tile[ty + j][tx] = in[(size_t)(bm + ty + j) * N + bn + tx];
  __syncthreads();
#pragma unroll
  for (int j = 0; j < 32; j += 8)
    out[(size_t)(bn + ty + j) * M + bm + tx] = (__bf16)tile[tx][ty + j];
}

// ---------------------------------------------------------------------------
// LayerNorm: one wave32 per row of 256, bf16 output
// ---------------------------------------------------------------------------
__global__ __launch_bounds__(128)
void ln_bf16_kernel(const float* __restrict__ x, const float* __restrict__ w,
                    const float* __restrict__ b, __bf16* __restrict__ out) {
  int row  = blockIdx.x * 4 + (threadIdx.x >> 5);
  int lane = threadIdx.x & 31;
  const float* xr = x + (size_t)row * CC + lane * 8;
  float v[8];
#pragma unroll
  for (int i = 0; i < 8; ++i) v[i] = xr[i];
  float s = 0.f;
#pragma unroll
  for (int i = 0; i < 8; ++i) s += v[i];
#pragma unroll
  for (int m = 1; m < 32; m <<= 1) s += __shfl_xor(s, m, 32);
  float mean = s * (1.0f / CC);
  float vs = 0.f;
#pragma unroll
  for (int i = 0; i < 8; ++i) { float d = v[i] - mean; vs += d * d; }
#pragma unroll
  for (int m = 1; m < 32; m <<= 1) vs += __shfl_xor(vs, m, 32);
  float rstd = rsqrtf(vs * (1.0f / CC) + LN_EPS);
  __bf16* orow = out + (size_t)row * CC + lane * 8;
#pragma unroll
  for (int i = 0; i < 8; ++i) {
    int col = lane * 8 + i;
    orow[i] = (__bf16)((v[i] - mean) * rstd * w[col] + b[col]);
  }
}

// ---------------------------------------------------------------------------
// RoPE: out[2p+i] = f[i][0]*x[2p] + f[i][1]*x[2p+1]; Q gets softmax scale folded
// ---------------------------------------------------------------------------
__global__ void rope_q_kernel(const float* __restrict__ q,
                              const float* __restrict__ fr,
                              __bf16* __restrict__ out, float scale, int rows) {
  int i = blockIdx.x * 256 + threadIdx.x;
  if (i >= rows * 128) return;
  int p = i >> 7, d2 = i & 127;
  const float* f = fr + ((size_t)p * 128 + d2) * 4;
  float x0 = q[(size_t)p * CC + 2 * d2], x1 = q[(size_t)p * CC + 2 * d2 + 1];
  out[(size_t)p * CC + 2 * d2]     = (__bf16)((f[0] * x0 + f[1] * x1) * scale);
  out[(size_t)p * CC + 2 * d2 + 1] = (__bf16)((f[2] * x0 + f[3] * x1) * scale);
}

__global__ void rope_k_kernel(const float* __restrict__ k,
                              const float* __restrict__ fr,
                              const int* __restrict__ nkx_ptr,
                              __bf16* __restrict__ out, int rows, int nf) {
  int i = blockIdx.x * 256 + threadIdx.x;
  if (i >= rows * 128) return;
  int p = i >> 7, d2 = i & 127;
  int nkr = rows - (nkx_ptr ? *nkx_ptr : 0);
  float x0 = k[(size_t)p * CC + 2 * d2], x1 = k[(size_t)p * CC + 2 * d2 + 1];
  float o0 = x0, o1 = x1;
  if (p < nkr) {
    const float* f = fr + ((size_t)(p % nf) * 128 + d2) * 4;
    o0 = f[0] * x0 + f[1] * x1;
    o1 = f[2] * x0 + f[3] * x1;
  }
  out[(size_t)p * CC + 2 * d2]     = (__bf16)o0;
  out[(size_t)p * CC + 2 * d2 + 1] = (__bf16)o1;
}

// ---------------------------------------------------------------------------
// Generic bf16 WMMA GEMM: D[M][N] = A[M][K] * B[N][K]^T + bias (+ addsrc)
// block = 128 thr (4 waves); block tile 64x64; wave tile 32x32 (2x2 WMMA).
// Fused epilogue: residual add, exact GELU, fp32 or bf16 output.
// ---------------------------------------------------------------------------
__global__ __launch_bounds__(128)
void gemm_bf16_kernel(const __bf16* __restrict__ A, const __bf16* __restrict__ B,
                      const float* __restrict__ bias,
                      const float* __restrict__ addsrc,
                      void* __restrict__ out, int M, int N, int K,
                      int outBf16, int doGelu) {
  const int lane = threadIdx.x & 31;
  const int wave = threadIdx.x >> 5;
  const int hl   = lane >> 4;      // half-wave select (K offset 0 / 8)
  const int l15  = lane & 15;
  const int Mbase = blockIdx.y * 64 + (wave >> 1) * 32;
  const int Nbase = blockIdx.x * 64 + (wave & 1) * 32;

  v8f acc[2][2];
#pragma unroll
  for (int a = 0; a < 2; ++a)
#pragma unroll
    for (int c = 0; c < 2; ++c)
#pragma unroll
      for (int r = 0; r < 8; ++r) acc[a][c][r] = 0.f;

  const int koff = hl * 8;
  const __bf16* a0p = A + (size_t)(Mbase + l15) * K + koff;
  const __bf16* a1p = A + (size_t)(Mbase + 16 + l15) * K + koff;
  const __bf16* b0p = B + (size_t)(Nbase + l15) * K + koff;
  const __bf16* b1p = B + (size_t)(Nbase + 16 + l15) * K + koff;

  for (int kb = 0; kb < K; kb += 32) {
    if (kb + 32 < K) {   // WGP-scope global_prefetch_b8 of next K chunk
      __builtin_prefetch(a0p + kb + 32, 0, 3);
      __builtin_prefetch(b0p + kb + 32, 0, 3);
    }
    v16bf a0 = load_frag(a0p + kb);
    v16bf a1 = load_frag(a1p + kb);
    v16bf b0 = load_frag(b0p + kb);
    v16bf b1 = load_frag(b1p + kb);
    acc[0][0] = wmma_bf16(a0, b0, acc[0][0]);
    acc[0][1] = wmma_bf16(a0, b1, acc[0][1]);
    acc[1][0] = wmma_bf16(a1, b0, acc[1][0]);
    acc[1][1] = wmma_bf16(a1, b1, acc[1][1]);
  }

#pragma unroll
  for (int mi = 0; mi < 2; ++mi)
#pragma unroll
    for (int ni = 0; ni < 2; ++ni)
#pragma unroll
      for (int r = 0; r < 8; ++r) {
        int row = Mbase + mi * 16 + r + hl * 8;   // C/D layout: VGPR r -> M=r / M=8+r
        int col = Nbase + ni * 16 + l15;
        float v = acc[mi][ni][r] + bias[col];
        if (addsrc) v += addsrc[(size_t)row * N + col];
        if (doGelu) v = 0.5f * v * (1.0f + erff(v * 0.70710678118654752f));
        if (outBf16) ((__bf16*)out)[(size_t)row * N + col] = (__bf16)v;
        else         ((float*)out)[(size_t)row * N + col] = v;
      }
}

// ---------------------------------------------------------------------------
// Split-K flash attention (wave32 WMMA). Each wave: 16 queries, full d=256.
// K/V tiles for each 32-key chunk are staged in LDS ONCE per block and shared
// by all 4 waves (4x traffic reduction on the dominant stream). LDS rows are
// padded (264 / 40 elems) so fragment reads hit distinct banks.
// Processes keys [kb0, kb1); emits UNNORMALIZED O plus per-row (m, l).
// Q is pre-scaled by 1/16.
// ---------------------------------------------------------------------------
__global__ __launch_bounds__(128)
void flash_attn_kernel(const __bf16* __restrict__ Q, const __bf16* __restrict__ Kb,
                       const __bf16* __restrict__ VT,
                       float* __restrict__ Opart, float* __restrict__ mpart,
                       float* __restrict__ lpart, int NKv, int kb0, int kb1) {
  __shared__ __bf16 ktile[32][264];   // 32 keys x 256 d (padded)
  __shared__ __bf16 vtile[256][40];   // 256 d x 32 keys (padded)
  __shared__ __bf16 plds[4][16 * 32];
  const int tid   = threadIdx.x;
  const int lane  = tid & 31;
  const int wave  = tid >> 5;
  const int hl    = lane >> 4;
  const int l15   = lane & 15;
  const int koff  = hl * 8;
  const int qbase = blockIdx.x * 64 + wave * 16;

  // Preload the wave's Q as 8 A-fragments (d = 256 in chunks of 32)
  v16bf qf[8];
  {
    const __bf16* qrow = Q + (size_t)(qbase + l15) * CC + koff;
#pragma unroll
    for (int c = 0; c < 8; ++c) qf[c] = load_frag(qrow + c * 32);
  }

  v8f o[16];
#pragma unroll
  for (int t = 0; t < 16; ++t)
#pragma unroll
    for (int r = 0; r < 8; ++r) o[t][r] = 0.f;

  float mrun[8], lrun[8];
#pragma unroll
  for (int r = 0; r < 8; ++r) { mrun[r] = -INFINITY; lrun[r] = 0.f; }

  __bf16* pl = plds[wave];

  for (int kbk = kb0; kbk < kb1; kbk += 32) {
    __syncthreads();   // all waves done reading previous K/V tiles
    // stage K tile: 32 rows x 256 (1024 x 16B units, 8 per thread, coalesced)
#pragma unroll
    for (int it = 0; it < 8; ++it) {
      int u = tid + it * 128;
      int r = u >> 5, cu = u & 31;
      *(v8bf*)&ktile[r][cu * 8] =
          *(const v8bf*)(Kb + (size_t)(kbk + r) * CC + cu * 8);
    }
    // stage V tile: 256 d-rows x 32 keys (1024 x 16B units)
#pragma unroll
    for (int it = 0; it < 8; ++it) {
      int u = tid + it * 128;
      int dc = u >> 2, cu = u & 3;
      *(v8bf*)&vtile[dc][cu * 8] =
          *(const v8bf*)(VT + (size_t)dc * NKv + kbk + cu * 8);
    }
    __syncthreads();

    // S = Q * K^T for 32 keys (two 16x16 tiles), K from LDS
    v8f s0, s1;
#pragma unroll
    for (int r = 0; r < 8; ++r) { s0[r] = 0.f; s1[r] = 0.f; }
#pragma unroll
    for (int c = 0; c < 8; ++c) {
      v16bf b0 = load_frag(&ktile[l15][c * 32 + koff]);
      v16bf b1 = load_frag(&ktile[16 + l15][c * 32 + koff]);
      s0 = wmma_bf16(qf[c], b0, s0);
      s1 = wmma_bf16(qf[c], b1, s1);
    }

    // online softmax (row stats live in the 16-lane half that owns the row)
    float pscale[8];
#pragma unroll
    for (int r = 0; r < 8; ++r) {
      float mx = fmaxf(s0[r], s1[r]);
#pragma unroll
      for (int m = 1; m < 16; m <<= 1) mx = fmaxf(mx, __shfl_xor(mx, m, 16));
      float mn = fmaxf(mrun[r], mx);
      float sc = __expf(mrun[r] - mn);
      float p0 = __expf(s0[r] - mn);
      float p1 = __expf(s1[r] - mn);
      s0[r] = p0; s1[r] = p1;
      float rs = p0 + p1;
#pragma unroll
      for (int m = 1; m < 16; m <<= 1) rs += __shfl_xor(rs, m, 16);
      lrun[r] = lrun[r] * sc + rs;
      mrun[r] = mn;
      pscale[r] = sc;
    }
#pragma unroll
    for (int t = 0; t < 16; ++t)
#pragma unroll
      for (int r = 0; r < 8; ++r) o[t][r] *= pscale[r];

    // P: accumulator layout -> LDS (row-major 16x32) -> A-fragment layout
#pragma unroll
    for (int r = 0; r < 8; ++r) {
      int m = r + hl * 8;
      pl[m * 32 + l15]      = (__bf16)s0[r];
      pl[m * 32 + 16 + l15] = (__bf16)s1[r];
    }
    __syncthreads();
    v16bf ap;
    {
      const __bf16* p = pl + l15 * 32 + koff;
#pragma unroll
      for (int i = 0; i < 8; ++i) { ap[i] = p[i]; ap[8 + i] = p[16 + i]; }
    }

    // O += P * V, V from LDS (vtile row = d column; keys contiguous)
#pragma unroll
    for (int t = 0; t < 16; ++t) {
      v16bf bv = load_frag(&vtile[t * 16 + l15][koff]);
      o[t] = wmma_bf16(ap, bv, o[t]);
    }
  }

#pragma unroll
  for (int t = 0; t < 16; ++t)
#pragma unroll
    for (int r = 0; r < 8; ++r) {
      int m = r + hl * 8;
      Opart[(size_t)(qbase + m) * CC + t * 16 + l15] = o[t][r];
    }
  if (l15 == 0) {
#pragma unroll
    for (int r = 0; r < 8; ++r) {
      int m = r + hl * 8;
      mpart[qbase + m] = mrun[r];
      lpart[qbase + m] = lrun[r];
    }
  }
}

// Merge split-K partials: O = sum_s O_s * exp(m_s - M) / (sum_s l_s * exp(m_s - M))
__global__ __launch_bounds__(128)
void flash_combine_kernel(const float* __restrict__ Opart,
                          const float* __restrict__ mpart,
                          const float* __restrict__ lpart,
                          float* __restrict__ O, int nsplit) {
  int row  = blockIdx.x * 4 + (threadIdx.x >> 5);
  int lane = threadIdx.x & 31;
  float mg = -INFINITY;
  for (int s = 0; s < nsplit; ++s) mg = fmaxf(mg, mpart[(size_t)s * NQ + row]);
  float e[CA_SPLIT];
  float L = 0.f;
  for (int s = 0; s < nsplit; ++s) {
    e[s] = __expf(mpart[(size_t)s * NQ + row] - mg);
    L += lpart[(size_t)s * NQ + row] * e[s];
  }
  float rl = 1.0f / L;
#pragma unroll
  for (int i = 0; i < 8; ++i) {
    int col = lane * 8 + i;
    float acc = 0.f;
    for (int s = 0; s < nsplit; ++s)
      acc += Opart[((size_t)s * NQ + row) * CC + col] * e[s];
    O[(size_t)row * CC + col] = acc * rl;
  }
}

// ---------------------------------------------------------------------------
// Host orchestration
// ---------------------------------------------------------------------------
extern "C" void kernel_launch(void* const* d_in, const int* in_sizes, int n_in,
                              void* d_out, int out_size, void* d_ws, size_t ws_size,
                              hipStream_t stream) {
  (void)in_sizes; (void)n_in; (void)out_size; (void)ws_size;

  const float* image   = (const float*)d_in[0];
  const float* x_in    = (const float*)d_in[1];
  const float* mem_img = (const float*)d_in[2];
  const float* memory  = (const float*)d_in[3];
  const float* mem_pos = (const float*)d_in[4];
  const float* rope    = (const float*)d_in[5];
  const int*   nkx     = (const int*)d_in[6];
  // 0:sa_q 1:sa_k 2:sa_v 3:sa_o 4:ca_q 5:ca_k 6:ca_v 7:ca_o 8:ica_q 9:ica_k
  const float *W[10], *Bv[10];
  for (int i = 0; i < 10; ++i) { W[i] = (const float*)d_in[7 + 2 * i]; Bv[i] = (const float*)d_in[8 + 2 * i]; }
  const float* l1w = (const float*)d_in[27]; const float* l1b = (const float*)d_in[28];
  const float* l2w = (const float*)d_in[29]; const float* l2b = (const float*)d_in[30];
  const float* n1w = (const float*)d_in[31]; const float* n1b = (const float*)d_in[32];
  const float* n2w = (const float*)d_in[33]; const float* n2b = (const float*)d_in[34];
  const float* n3w = (const float*)d_in[35]; const float* n3b = (const float*)d_in[36];

  size_t off = 0; char* base = (char*)d_ws;
  auto alloc = [&](size_t bytes) -> void* {
    void* p = base + off; off += (bytes + 255) & ~(size_t)255; return p;
  };

  __bf16* wb[10];
  for (int i = 0; i < 10; ++i) wb[i] = (__bf16*)alloc((size_t)CC * CC * 2);
  __bf16* l1wb    = (__bf16*)alloc((size_t)FF * CC * 2);
  __bf16* l2wb    = (__bf16*)alloc((size_t)CC * FF * 2);
  __bf16* normed  = (__bf16*)alloc((size_t)NQ * CC * 2);
  __bf16* image_b = (__bf16*)alloc((size_t)NQ * CC * 2);
  __bf16* mimg_b  = (__bf16*)alloc((size_t)NKK * CC * 2);
  __bf16* mem_b   = (__bf16*)alloc((size_t)NKK * CC * 2);
  float*  f32q    = (float*)alloc((size_t)NKK * CC * 4);
  float*  f32k    = (float*)alloc((size_t)NKK * CC * 4);
  float*  f32v    = (float*)alloc((size_t)NKK * CC * 4);
  __bf16* qb      = (__bf16*)alloc((size_t)NQ * CC * 2);
  __bf16* kb_     = (__bf16*)alloc((size_t)NKK * CC * 2);
  __bf16* vT      = (__bf16*)alloc((size_t)CC * NKK * 2);
  float*  opart   = (float*)alloc((size_t)CA_SPLIT * NQ * CC * 4);
  float*  mpart   = (float*)alloc((size_t)CA_SPLIT * NQ * 4);
  float*  lpart   = (float*)alloc((size_t)CA_SPLIT * NQ * 4);
  float*  obuf    = (float*)alloc((size_t)NQ * CC * 4);
  __bf16* ob16    = (__bf16*)alloc((size_t)NQ * CC * 2);
  float*  xb1     = (float*)alloc((size_t)NQ * CC * 4);
  float*  xb2     = (float*)alloc((size_t)NQ * CC * 4);
  __bf16* hidden  = (__bf16*)alloc((size_t)NQ * FF * 2);

  auto cvt = [&](const float* in, __bf16* out, int n) {
    cvt_bf16_kernel<<<(n / 4 + 255) / 256, 256, 0, stream>>>(in, out, n);
  };
  auto gemm = [&](const __bf16* A, const __bf16* Bm, const float* bias,
                  const float* addsrc, void* out, int M, int N, int K,
                  int outb, int gelu) {
    dim3 g(N / 64, M / 64);
    gemm_bf16_kernel<<<g, 128, 0, stream>>>(A, Bm, bias, addsrc, out, M, N, K, outb, gelu);
  };
  auto attn = [&](const __bf16* Qp, const __bf16* Kp, const __bf16* Vp,
                  float* Out, int NKv, int nsplit) {
    int chunks = NKv / 32;
    int per = (chunks + nsplit - 1) / nsplit;            // chunks per split
    for (int s = 0; s < nsplit; ++s) {
      int b0 = s * per * 32;
      int b1 = (s + 1) * per * 32; if (b1 > NKv) b1 = NKv;
      flash_attn_kernel<<<NQ / 64, 128, 0, stream>>>(
          Qp, Kp, Vp, opart + (size_t)s * NQ * CC,
          mpart + (size_t)s * NQ, lpart + (size_t)s * NQ, NKv, b0, b1);
    }
    flash_combine_kernel<<<NQ / 4, 128, 0, stream>>>(opart, mpart, lpart, Out, nsplit);
  };
  auto transpose = [&](const float* in, __bf16* out, int M, int N) {
    dim3 g(N / 32, M / 32);
    transpose_cvt_kernel<<<g, 256, 0, stream>>>(in, out, M, N);
  };

  // weight / input conversions to bf16
  for (int i = 0; i < 10; ++i) cvt(W[i], wb[i], CC * CC);
  cvt(l1w, l1wb, FF * CC);
  cvt(l2w, l2wb, CC * FF);
  cvt(image, image_b, NQ * CC);
  cvt(mem_img, mimg_b, NKK * CC);
  cvt(memory, mem_b, NKK * CC);

  const float scale = 0.0625f; // 256^-0.5

  // ---- self attention ----
  ln_bf16_kernel<<<NQ / 4, 128, 0, stream>>>(x_in, n1w, n1b, normed);
  gemm(normed, wb[0], Bv[0], nullptr, f32q, NQ, CC, CC, 0, 0);
  gemm(normed, wb[1], Bv[1], nullptr, f32k, NQ, CC, CC, 0, 0);
  gemm(normed, wb[2], Bv[2], nullptr, f32v, NQ, CC, CC, 0, 0);
  rope_q_kernel<<<(NQ * 128 + 255) / 256, 256, 0, stream>>>(f32q, rope, qb, scale, NQ);
  rope_k_kernel<<<(NQ * 128 + 255) / 256, 256, 0, stream>>>(f32k, rope, nullptr, kb_, NQ, NQ);
  transpose(f32v, vT, NQ, CC);
  attn(qb, kb_, vT, obuf, NQ, SA_SPLIT);
  cvt(obuf, ob16, NQ * CC);
  gemm(ob16, wb[3], Bv[3], x_in, xb1, NQ, CC, CC, 0, 0); // x = x + proj

  // ---- cross attention ----
  ln_bf16_kernel<<<NQ / 4, 128, 0, stream>>>(xb1, n2w, n2b, normed);
  gemm(image_b, wb[8], Bv[8], nullptr, f32q, NQ, CC, CC, 0, 0);   // ica_q(image)
  gemm(normed,  wb[4], Bv[4], f32q,    f32q, NQ, CC, CC, 0, 0);   // + ca_q(normed)
  gemm(mimg_b,  wb[9], Bv[9], mem_pos, f32k, NKK, CC, CC, 0, 0);  // ica_k + pos
  gemm(mem_b,   wb[5], Bv[5], f32k,    f32k, NKK, CC, CC, 0, 0);  // + ca_k
  gemm(mem_b,   wb[6], Bv[6], nullptr, f32v, NKK, CC, CC, 0, 0);  // ca_v
  rope_q_kernel<<<(NQ * 128 + 255) / 256, 256, 0, stream>>>(f32q, rope, qb, scale, NQ);
  rope_k_kernel<<<(NKK * 128 + 255) / 256, 256, 0, stream>>>(f32k, rope, nkx, kb_, NKK, NQ);
  transpose(f32v, vT, NKK, CC);
  attn(qb, kb_, vT, obuf, NKK, CA_SPLIT);
  cvt(obuf, ob16, NQ * CC);
  gemm(ob16, wb[7], Bv[7], xb1, xb2, NQ, CC, CC, 0, 0);

  // ---- MLP ----
  ln_bf16_kernel<<<NQ / 4, 128, 0, stream>>>(xb2, n3w, n3b, normed);
  gemm(normed, l1wb, l1b, nullptr, hidden, NQ, FF, CC, /*bf16*/1, /*gelu*/1);
  float* xout = (float*)d_out + (size_t)NQ * CC;
  gemm(hidden, l2wb, l2b, xb2, xout, NQ, CC, FF, 0, 0);

  // image passes through unchanged
  hipMemcpyAsync(d_out, image, (size_t)NQ * CC * sizeof(float),
                 hipMemcpyDeviceToDevice, stream);
}
